// MultiHeadSelfAttention_8761733284269
// MI455X (gfx1250) — matrix operand order repaired and checked by reading, not compiled, not run on hardware
//
#include <hip/hip_runtime.h>
#include <hip/hip_bf16.h>
#include <stdint.h>

#define DIM   4096
#define BATCH 8192
#define EPS   1e-5f

// ---- WMMA vector types (CDNA5 / gfx1250, wave32) ----
typedef __attribute__((ext_vector_type(16))) __bf16 v16bf;
typedef __attribute__((ext_vector_type(8)))  float  v8f;

// ---- GEMM tiling ----
#define BM   128    // workgroup M tile
#define BN   128    // workgroup N tile
#define BK   32     // K chunk = one v_wmma_f32_16x16x32_bf16 step
#define LDSS 40     // LDS row stride in bf16 elems (padded vs 32 -> conflict-free b128)

// gfx1250 async global->LDS path (ASYNCcnt), gated on toolchain support
#if defined(__gfx1250__) && __has_builtin(__builtin_amdgcn_global_load_async_to_lds_b128)
#define HAVE_ASYNC_LDS 1
#else
#define HAVE_ASYNC_LDS 0
#endif

// Exact parameter types per hipcc diagnostic:
//   param 0: 'int __attribute__((vector_size(16))) addrspace(1)*'  (global int4)
//   param 1: LDS-side pointer (addrspace(3)), same vector pointee
typedef int v4i_n __attribute__((vector_size(4 * sizeof(int))));
typedef __attribute__((address_space(1))) v4i_n gv4i;
typedef __attribute__((address_space(3))) v4i_n lv4i;

__device__ __forceinline__ unsigned short f32_bf16(float f) {
    unsigned int u = __float_as_uint(f);
    u += 0x7FFFu + ((u >> 16) & 1u);          // round-to-nearest-even
    return (unsigned short)(u >> 16);
}

// pack 8 fp32 (two uint4 views) -> 8 bf16 in one uint4
__device__ __forceinline__ uint4 cvt8_bf16(uint4 q0, uint4 q1) {
    union { uint4 q; float f[4]; } a, b;
    a.q = q0; b.q = q1;
    uint4 r;
    r.x = (unsigned)f32_bf16(a.f[0]) | ((unsigned)f32_bf16(a.f[1]) << 16);
    r.y = (unsigned)f32_bf16(a.f[2]) | ((unsigned)f32_bf16(a.f[3]) << 16);
    r.z = (unsigned)f32_bf16(b.f[0]) | ((unsigned)f32_bf16(b.f[1]) << 16);
    r.w = (unsigned)f32_bf16(b.f[2]) | ((unsigned)f32_bf16(b.f[3]) << 16);
    return r;
}

__device__ __forceinline__ void async_copy_b128(const unsigned short* g,
                                                unsigned short* l) {
#if HAVE_ASYNC_LDS
    __builtin_amdgcn_global_load_async_to_lds_b128(
        (gv4i*)g, (lv4i*)l, /*imm offset*/0, /*cpol*/0);
#else
    (void)g; (void)l;
#endif
}

__device__ __forceinline__ void wait_asynccnt0() {
#if defined(__gfx1250__)
#if __has_builtin(__builtin_amdgcn_s_wait_asynccnt)
    __builtin_amdgcn_s_wait_asynccnt(0);
#else
    asm volatile("s_wait_asynccnt 0x0" ::: "memory");
#endif
#endif
}

union FragU { uint4 q[2]; v16bf v; };

// ---------------------------------------------------------------------------
// Elementwise fp32 -> bf16 (bandwidth bound pre-pass); n multiple of 2048
// ---------------------------------------------------------------------------
__global__ __launch_bounds__(256)
void cvt_f32_bf16(const float* __restrict__ in, unsigned short* __restrict__ out) {
    const size_t i = ((size_t)blockIdx.x * 256 + threadIdx.x) * 8;
    uint4 a = *(const uint4*)(in + i);
    uint4 b = *(const uint4*)(in + i + 4);
    *(uint4*)(out + i) = cvt8_bf16(a, b);
}

// ---------------------------------------------------------------------------
// C[M,N] = A[M,K](bf16) * W[N,K](bf16)^T + bias[N]
// Double-buffered LDS, async global->LDS fills, 8 WMMAs / K-chunk / wave.
// ---------------------------------------------------------------------------
template<bool OUT_BF16>
__global__ __launch_bounds__(256)
void gemm_bf16_wmma(const unsigned short* __restrict__ A,
                    const unsigned short* __restrict__ Bw,
                    const float* __restrict__ bias,
                    void* __restrict__ Outv) {
    constexpr int K = DIM, N = DIM;

    __shared__ __align__(16) unsigned short sA[2][BM * LDSS];
    __shared__ __align__(16) unsigned short sB[2][BN * LDSS];

    const int tid = threadIdx.x;
    const int m0  = blockIdx.y * BM;
    const int n0  = blockIdx.x * BN;

    // wave grid: 2 (M) x 4 (N); each wave -> 64x32 = 4x2 WMMA tiles of 16x16
    const int wave = tid >> 5;
    const int lane = tid & 31;
    const int lm   = lane & 15;
    const int lh   = lane >> 4;
    const int wm0  = (wave >> 2) * 64;
    const int wn0  = (wave & 3)  * 32;

    // loader: thread t stages 16 bf16 (32 B) of one row of A and of W
    const int lrow = tid >> 1;
    const int lcol = (tid & 1) * 16;

    const unsigned short* gA = A  + (size_t)(m0 + lrow) * K + lcol;
    const unsigned short* gB = Bw + (size_t)(n0 + lrow) * K + lcol;

    v8f acc[4][2];
    const v8f vzero = {0.f,0.f,0.f,0.f,0.f,0.f,0.f,0.f};
#pragma unroll
    for (int mt = 0; mt < 4; ++mt)
#pragma unroll
        for (int nt = 0; nt < 2; ++nt) acc[mt][nt] = vzero;

    auto stage = [&](int kt, int buf) {
        const unsigned short* pa = gA + kt * BK;
        const unsigned short* pb = gB + kt * BK;
        unsigned short* da = &sA[buf][lrow * LDSS + lcol];
        unsigned short* db = &sB[buf][lrow * LDSS + lcol];
#if HAVE_ASYNC_LDS
        async_copy_b128(pa,     da);
        async_copy_b128(pa + 8, da + 8);
        async_copy_b128(pb,     db);
        async_copy_b128(pb + 8, db + 8);
#else
        uint4 a0 = *(const uint4*)(pa);
        uint4 a1 = *(const uint4*)(pa + 8);
        uint4 b0 = *(const uint4*)(pb);
        uint4 b1 = *(const uint4*)(pb + 8);
        *(uint4*)(da)     = a0;
        *(uint4*)(da + 8) = a1;
        *(uint4*)(db)     = b0;
        *(uint4*)(db + 8) = b1;
#endif
        // L2 prefetch hint a couple of chunks ahead (global_prefetch_b8)
        __builtin_prefetch((const void*)(pb + 2 * BK), 0, 1);
        __builtin_prefetch((const void*)(pa + 2 * BK), 0, 1);
    };

    stage(0, 0);
    wait_asynccnt0();
    __syncthreads();

    const int NK = K / BK;
    for (int kt = 0; kt < NK; ++kt) {
        const int  cur  = kt & 1;
        const bool more = (kt + 1) < NK;
        if (more) stage(kt + 1, cur ^ 1);      // async fill of next buffer

        // A frag: 16x32 bf16.  Lane: M=lm, K = {lh*8..+8} U {16+lh*8..+8}
        v16bf afrag[4], bfrag[2];
#pragma unroll
        for (int mt = 0; mt < 4; ++mt) {
            const unsigned short* pa = &sA[cur][(wm0 + mt * 16 + lm) * LDSS + lh * 8];
            FragU u;
            u.q[0] = *(const uint4*)(pa);
            u.q[1] = *(const uint4*)(pa + 16);
            afrag[mt] = u.v;
        }
        // B frag: 32x16 bf16.  Lane: N=lm, K = {lh*8..+8} U {16+lh*8..+8}
#pragma unroll
        for (int nt = 0; nt < 2; ++nt) {
            const unsigned short* pb = &sB[cur][(wn0 + nt * 16 + lm) * LDSS + lh * 8];
            FragU u;
            u.q[0] = *(const uint4*)(pb);
            u.q[1] = *(const uint4*)(pb + 16);
            bfrag[nt] = u.v;
        }

#pragma unroll
        for (int mt = 0; mt < 4; ++mt)
#pragma unroll
            for (int nt = 0; nt < 2; ++nt)
                acc[mt][nt] = __builtin_amdgcn_wmma_f32_16x16x32_bf16(
                    false, afrag[mt], false, bfrag[nt],
                    (short)0, acc[mt][nt], false, false);

        if (more) wait_asynccnt0();            // next buffer fully in LDS
        __syncthreads();
    }

    // Epilogue.  C/D layout: lane -> N = lm, VGPR r -> M = lh*8 + r
#pragma unroll
    for (int mt = 0; mt < 4; ++mt)
#pragma unroll
        for (int nt = 0; nt < 2; ++nt) {
            const int gm = m0 + wm0 + mt * 16 + lh * 8;
            const int gn = n0 + wn0 + nt * 16 + lm;
            const float bs = bias[gn];
#pragma unroll
            for (int r = 0; r < 8; ++r) {
                const float val = acc[mt][nt][r] + bs;
                if constexpr (OUT_BF16)
                    ((unsigned short*)Outv)[(size_t)(gm + r) * N + gn] = f32_bf16(val);
                else
                    ((float*)Outv)[(size_t)(gm + r) * N + gn] = val;
            }
        }
}

// ---------------------------------------------------------------------------
// In-place LayerNorm over rows of y[BATCH, DIM]; one 256-thread block per row
// ---------------------------------------------------------------------------
__global__ __launch_bounds__(256)
void layernorm_inplace(float* __restrict__ y,
                       const float* __restrict__ gamma,
                       const float* __restrict__ beta) {
    __shared__ float red[256];
    float* p = y + (size_t)blockIdx.x * DIM;
    const int c0 = threadIdx.x * 16;

    float v[16];
#pragma unroll
    for (int i = 0; i < 16; i += 4) {
        float4 t = *(const float4*)(p + c0 + i);
        v[i] = t.x; v[i+1] = t.y; v[i+2] = t.z; v[i+3] = t.w;
    }
    float s = 0.f, s2 = 0.f;
#pragma unroll
    for (int i = 0; i < 16; ++i) { s += v[i]; s2 += v[i] * v[i]; }

    red[threadIdx.x] = s;  __syncthreads();
    for (int off = 128; off > 0; off >>= 1) {
        if (threadIdx.x < off) red[threadIdx.x] += red[threadIdx.x + off];
        __syncthreads();
    }
    const float mean = red[0] * (1.0f / DIM);
    __syncthreads();
    red[threadIdx.x] = s2; __syncthreads();
    for (int off = 128; off > 0; off >>= 1) {
        if (threadIdx.x < off) red[threadIdx.x] += red[threadIdx.x + off];
        __syncthreads();
    }
    const float var = red[0] * (1.0f / DIM) - mean * mean;
    const float inv = rsqrtf(var + EPS);

#pragma unroll
    for (int i = 0; i < 16; i += 4) {
        float4 o;
        o.x = (v[i+0] - mean) * inv * gamma[c0+i+0] + beta[c0+i+0];
        o.y = (v[i+1] - mean) * inv * gamma[c0+i+1] + beta[c0+i+1];
        o.z = (v[i+2] - mean) * inv * gamma[c0+i+2] + beta[c0+i+2];
        o.w = (v[i+3] - mean) * inv * gamma[c0+i+3] + beta[c0+i+3];
        *(float4*)(p + c0 + i) = o;
    }
}

extern "C" void kernel_launch(void* const* d_in, const int* in_sizes, int n_in,
                              void* d_out, int out_size, void* d_ws, size_t ws_size,
                              hipStream_t stream) {
    const float* x      = (const float*)d_in[0];
    const float* W_qkv  = (const float*)d_in[1];
    const float* b_qkv  = (const float*)d_in[2];
    const float* W_proj = (const float*)d_in[3];
    const float* b_proj = (const float*)d_in[4];
    const float* gamma  = (const float*)d_in[5];
    const float* beta   = (const float*)d_in[6];
    float* out = (float*)d_out;

    // softmax over a length-1 axis == 1  =>  attention output == V.
    // Only the V third of W_qkv/b_qkv is ever needed.
    const float* Wv = W_qkv + (size_t)2 * DIM * DIM;
    const float* bv = b_qkv + 2 * DIM;

    // workspace layout (bf16 = ushort):
    //   xh  [BATCH*DIM]  64 MB
    //   Wvh [DIM*DIM]    32 MB
    //   Wph [DIM*DIM]    32 MB
    //   V   [BATCH*DIM]  64 MB            total 192 MB
    unsigned short* xh  = (unsigned short*)d_ws;
    unsigned short* Wvh = xh  + (size_t)BATCH * DIM;
    unsigned short* Wph = Wvh + (size_t)DIM * DIM;
    unsigned short* V   = Wph + (size_t)DIM * DIM;

    dim3 block(256);

    // pre-pass: everything to bf16 once (removes all conversion from GEMM loops)
    cvt_f32_bf16<<<dim3((BATCH * (size_t)DIM) / 2048), block, 0, stream>>>(x, xh);
    cvt_f32_bf16<<<dim3(((size_t)DIM * DIM) / 2048), block, 0, stream>>>(Wv, Wvh);
    cvt_f32_bf16<<<dim3(((size_t)DIM * DIM) / 2048), block, 0, stream>>>(W_proj, Wph);

    dim3 grid(DIM / BN, BATCH / BM);
    // V = x @ Wv^T + bv
    gemm_bf16_wmma<true ><<<grid, block, 0, stream>>>(xh, Wvh, bv, V);
    // out = V @ W_proj^T + b_proj
    gemm_bf16_wmma<false><<<grid, block, 0, stream>>>(V, Wph, b_proj, out);
    // out = LN(out) * gamma + beta (in place)
    layernorm_inplace<<<dim3(BATCH), block, 0, stream>>>(out, gamma, beta);
}
